// Attention_22368189678325
// MI455X (gfx1250) — matrix operand order repaired and checked by reading, not compiled, not run on hardware
//
#include <hip/hip_runtime.h>
#include <hip/hip_bf16.h>

typedef __attribute__((ext_vector_type(16))) __bf16 v16bf;
typedef __attribute__((ext_vector_type(8)))  float  v8f;

#define SEQ      32
#define NEMBD    1024
#define NHEAD    16
#define HDIM     64
#define PASTLEN  32768
#define TOTLEN   32800
#define NCH      65            // 64 past chunks of 512 keys + 1 new chunk of 32
#define CHUNK    512
#define KROW     68            // padded LDS row stride (floats), 272B = 17*16B
#define SROW     36            // padded LDS row stride for the P (score) tile
#define WSLICE   (2*32*KROW + 32*SROW)   // 5504 floats of LDS per wave
#define PARTSZ   2112          // per-(head,chunk) partial: m[32] l[32] acc[32*64]

static __device__ __forceinline__ v8f wmma_bf16(v16bf a, v16bf b, v8f c) {
  // D = A(16x32 bf16) * B(32x16 bf16) + C(16x16 f32)
  return __builtin_amdgcn_wmma_f32_16x16x32_bf16(false, a, false, b, (short)0, c,
                                                 false, false);
}

// Low 32 bits of a generic pointer to __shared__ == LDS byte offset on amdgcn
// (flat shared address = {SHARED_BASE_HI, lds_offset[31:0]}, ISA 10.2).
static __device__ __forceinline__ unsigned lds_off(const void* p) {
  return (unsigned)(unsigned long long)p;
}

// ---------------------------------------------------------------------------
// C[32,N] = A[32,K] * B[K,N] + bias[N], bf16 WMMA, one 16x16 tile per wave.
// block = 64 threads (2 waves: mt = wave id), grid.x = N/16.
// ---------------------------------------------------------------------------
__global__ __launch_bounds__(64)
void gemm32_bias_wmma(const float* __restrict__ A, const float* __restrict__ B,
                      const float* __restrict__ bias, float* __restrict__ C,
                      int K, int N) {
  const int lane = threadIdx.x & 31;
  const int l16  = lane & 15;
  const int lhi  = lane >> 4;
  const int mt   = threadIdx.x >> 5;   // 0..1 -> rows [mt*16, mt*16+16)
  const int nt   = blockIdx.x;
  const int m = mt * 16 + l16;         // A-layout: lane%16 = M row
  const int n = nt * 16 + l16;         // B/C-layout: lane%16 = N col
  v8f c = {};
  for (int k0 = 0; k0 < K; k0 += 32) {
    v16bf a, b;
#pragma unroll
    for (int j = 0; j < 16; ++j) {
      // A 16x32 bf16 layout: k = (j/8)*16 + (lane/16)*8 + j%8
      const int ka = k0 + ((j >> 3) << 4) + (lhi << 3) + (j & 7);
      a[j] = (__bf16)A[m * K + ka];
      // B 32x16 bf16 layout: k = (j/8)*16 + (lane/16)*8 + j%8
      const int kb = k0 + ((j >> 3) << 4) + (lhi << 3) + (j & 7);
      b[j] = (__bf16)B[kb * N + n];
    }
    c = wmma_bf16(a, b, c);
  }
  const float bv = bias[n];
#pragma unroll
  for (int r = 0; r < 8; ++r) {        // C layout: M = r + 8*(lane/16) (+16*mt)
    const int row = mt * 16 + r + 8 * lhi;
    C[row * N + n] = c[r] + bv;
  }
}

// ---------------------------------------------------------------------------
// Fused: KV-cache chunk -> present copy AND flash-attention partial, with the
// data movement done by the CDNA5 async-LDS engine:
//   global_load_async_to_lds_b128  (stage K/V tile, no VGPR round trip)
//   global_store_async_from_lds_b128 (present copy, overlapped with WMMA)
// One wave per (head, chunk). block = 64 (2 independent waves), grid = 520.
// ---------------------------------------------------------------------------
__global__ __launch_bounds__(64)
void attn_chunk(const float* __restrict__ past,   // [2,16,32768,64]
                const float* __restrict__ qkv,    // [32,3072] (q|k|v)
                float* __restrict__ present,      // [2,16,32800,64]
                float* __restrict__ part) {
  __shared__ float lds[2 * WSLICE];
  const int lane = threadIdx.x & 31;
  const int l16  = lane & 15;
  const int lhi  = lane >> 4;
  const int wv   = threadIdx.x >> 5;
  const int wid  = blockIdx.x * 2 + wv;
  const int h    = wid / NCH;
  const int ch   = wid % NCH;
  const bool isNew = (ch == NCH - 1);

  float* ldsK = lds + wv * WSLICE;
  float* ldsV = ldsK + 32 * KROW;
  float* ldsS = ldsV + 32 * KROW;

  // ---- Q fragments for this head: A-layout, 2 M-tiles x 2 k-steps over D=64
  v16bf qa[2][2];
#pragma unroll
  for (int mt = 0; mt < 2; ++mt)
#pragma unroll
    for (int g = 0; g < 2; ++g)
#pragma unroll
      for (int j = 0; j < 16; ++j) {
        const int dd = g * 32 + ((j >> 3) << 4) + (lhi << 3) + (j & 7);
        qa[mt][g][j] = (__bf16)qkv[(mt * 16 + l16) * (3 * NEMBD) + h * HDIM + dd];
      }

  // ---- online-softmax state (per lane, per (mt, r) row of its half)
  float mrow[2][8], lrow[2][8];
  v8f o[2][4];
#pragma unroll
  for (int mt = 0; mt < 2; ++mt) {
#pragma unroll
    for (int r = 0; r < 8; ++r) { mrow[mt][r] = -3.0e38f; lrow[mt][r] = 0.f; }
#pragma unroll
    for (int dt = 0; dt < 4; ++dt) { v8f z = {}; o[mt][dt] = z; }
  }

  const int nsteps = isNew ? 1 : (CHUNK / 32);
  for (int st = 0; st < nsteps; ++st) {
    const int tbase = isNew ? PASTLEN : ch * CHUNK + st * 32;

    const float* srcK; const float* srcV; int stride;
    if (isNew) {  // new k/v live in qkv rows (stride 3072)
      srcK = qkv + NEMBD     + h * HDIM;
      srcV = qkv + 2 * NEMBD + h * HDIM;
      stride = 3 * NEMBD;
    } else {
      srcK = past + (h * PASTLEN + tbase) * HDIM;
      srcV = past + ((NHEAD + h) * PASTLEN + tbase) * HDIM;
      stride = HDIM;
    }
    float* pK = present + (h * TOTLEN + tbase) * HDIM;
    float* pV = present + ((NHEAD + h) * TOTLEN + tbase) * HDIM;

    // -- fence LDS reuse: prior async stores must finish reading LDS, and
    //    prior WMMA-fragment ds_loads must have drained, before restaging.
    asm volatile("s_wait_asynccnt 0\n\ts_wait_dscnt 0" ::: "memory");

    // -- stage 32 keys x 64 dims of K and V straight into LDS (async engine)
#pragma unroll
    for (int i = 0; i < 16; ++i) {
      const int idx = lane + i * 32;                 // 0..511 float4 slots
      const int row = idx >> 4, c4 = (idx & 15) << 2;
      const float* gK = srcK + row * stride + c4;
      const float* gV = srcV + row * stride + c4;
      const unsigned lK = lds_off(ldsK + row * KROW + c4);
      const unsigned lV = lds_off(ldsV + row * KROW + c4);
      asm volatile("global_load_async_to_lds_b128 %0, %1, off"
                   :: "v"(lK), "v"(gK) : "memory");
      asm volatile("global_load_async_to_lds_b128 %0, %1, off"
                   :: "v"(lV), "v"(gV) : "memory");
    }
    asm volatile("s_wait_asynccnt 0" ::: "memory");  // tiles resident in LDS

    // -- present copy: LDS -> global via async engine, overlaps with compute
#pragma unroll
    for (int i = 0; i < 16; ++i) {
      const int idx = lane + i * 32;
      const int row = idx >> 4, c4 = (idx & 15) << 2;
      float* gpK = pK + row * HDIM + c4;
      float* gpV = pV + row * HDIM + c4;
      const unsigned lK = lds_off(ldsK + row * KROW + c4);
      const unsigned lV = lds_off(ldsV + row * KROW + c4);
      asm volatile("global_store_async_from_lds_b128 %0, %1, off"
                   :: "v"(gpK), "v"(lK) : "memory");
      asm volatile("global_store_async_from_lds_b128 %0, %1, off"
                   :: "v"(gpV), "v"(lV) : "memory");
    }

    // -- B-frags of K^T (contraction over d): two 16-key N-tiles, two d-steps
    v16bf kb[2][2];
#pragma unroll
    for (int nt = 0; nt < 2; ++nt)
#pragma unroll
      for (int g = 0; g < 2; ++g)
#pragma unroll
        for (int j = 0; j < 16; ++j) {
          const int dd = g * 32 + ((j >> 3) << 4) + (lhi << 3) + (j & 7);
          kb[nt][g][j] = (__bf16)ldsK[(nt * 16 + l16) * KROW + dd];
        }

    // -- S = Q * K^T  (scaled), C-layout: lane%16 = key, r+8*lhi = query row
    v8f s[2][2];
#pragma unroll
    for (int mt = 0; mt < 2; ++mt)
#pragma unroll
      for (int nt = 0; nt < 2; ++nt) {
        v8f z = {};
        v8f acc = wmma_bf16(qa[mt][0], kb[nt][0], z);
        s[mt][nt] = wmma_bf16(qa[mt][1], kb[nt][1], acc);
      }
#pragma unroll
    for (int mt = 0; mt < 2; ++mt)
#pragma unroll
      for (int nt = 0; nt < 2; ++nt)
#pragma unroll
        for (int r = 0; r < 8; ++r) {
          float v = s[mt][nt][r] * 0.125f;  // 1/sqrt(64)
          if (isNew) {                       // causal mask on the 32 new keys
            const int key = nt * 16 + l16;
            const int qrow = mt * 16 + r + 8 * lhi;
            if (key > qrow) v = -10000.0f;   // reference uses -1e4
          }
          s[mt][nt][r] = v;
        }

    // -- online softmax update (row reductions across 16 lanes of each half)
#pragma unroll
    for (int mt = 0; mt < 2; ++mt) {
      float nm[8];
#pragma unroll
      for (int r = 0; r < 8; ++r) {
        float t = fmaxf(s[mt][0][r], s[mt][1][r]);
        t = fmaxf(t, __shfl_xor(t, 1));
        t = fmaxf(t, __shfl_xor(t, 2));
        t = fmaxf(t, __shfl_xor(t, 4));
        t = fmaxf(t, __shfl_xor(t, 8));
        nm[r] = fmaxf(mrow[mt][r], t);
      }
#pragma unroll
      for (int r = 0; r < 8; ++r) {
        const float sc = __expf(mrow[mt][r] - nm[r]);
        mrow[mt][r] = nm[r];
        s[mt][0][r] = __expf(s[mt][0][r] - nm[r]);
        s[mt][1][r] = __expf(s[mt][1][r] - nm[r]);
        float rs = s[mt][0][r] + s[mt][1][r];
        rs += __shfl_xor(rs, 1);
        rs += __shfl_xor(rs, 2);
        rs += __shfl_xor(rs, 4);
        rs += __shfl_xor(rs, 8);
        lrow[mt][r] = lrow[mt][r] * sc + rs;
#pragma unroll
        for (int dt = 0; dt < 4; ++dt) o[mt][dt][r] *= sc;
      }
    }

    // -- transpose P (C-layout -> A-layout) through LDS
#pragma unroll
    for (int mt = 0; mt < 2; ++mt)
#pragma unroll
      for (int nt = 0; nt < 2; ++nt)
#pragma unroll
        for (int r = 0; r < 8; ++r)
          ldsS[(mt * 16 + r + 8 * lhi) * SROW + nt * 16 + l16] = s[mt][nt][r];
    asm volatile("s_wait_dscnt 0" ::: "memory");

    // -- O += P * V : A = P[16x32 keys], B = V[32 keys x 16 dims]
    v16bf vb[4];
#pragma unroll
    for (int dt = 0; dt < 4; ++dt)
#pragma unroll
      for (int j = 0; j < 16; ++j)
        vb[dt][j] = (__bf16)ldsV[(((j >> 3) << 4) + (lhi << 3) + (j & 7)) * KROW + dt * 16 + l16];
    v16bf pa[2];
#pragma unroll
    for (int mt = 0; mt < 2; ++mt)
#pragma unroll
      for (int j = 0; j < 16; ++j) {
        const int kk = ((j >> 3) << 4) + (lhi << 3) + (j & 7);
        pa[mt][j] = (__bf16)ldsS[(mt * 16 + l16) * SROW + kk];
      }
#pragma unroll
    for (int mt = 0; mt < 2; ++mt)
#pragma unroll
      for (int dt = 0; dt < 4; ++dt)
        o[mt][dt] = wmma_bf16(pa[mt], vb[dt], o[mt][dt]);
  }

  // ---- write chunk partial (m, l, unnormalized acc)
  float* pp = part + (h * NCH + ch) * PARTSZ;
  if (l16 == 0) {
#pragma unroll
    for (int mt = 0; mt < 2; ++mt)
#pragma unroll
      for (int r = 0; r < 8; ++r) {
        const int row = mt * 16 + r + 8 * lhi;
        pp[row] = mrow[mt][r];
        pp[32 + row] = lrow[mt][r];
      }
  }
#pragma unroll
  for (int mt = 0; mt < 2; ++mt)
#pragma unroll
    for (int dt = 0; dt < 4; ++dt)
#pragma unroll
      for (int r = 0; r < 8; ++r) {
        const int row = mt * 16 + r + 8 * lhi;
        const int d = dt * 16 + l16;
        pp[64 + row * HDIM + d] = o[mt][dt][r];
      }
  // S_ENDPGM performs an implicit wait-idle, draining pending async stores.
}

// ---------------------------------------------------------------------------
// Merge 65 partials per (head, query) with log-sum-exp; write merged heads
// into Am[32,1024]. grid = 16*32 blocks, block = 64 threads (one per dim d).
// ---------------------------------------------------------------------------
__global__ __launch_bounds__(64)
void reduce_attn(const float* __restrict__ part, float* __restrict__ Am) {
  const int h = blockIdx.x >> 5;
  const int q = blockIdx.x & 31;
  const int d = threadIdx.x;
  float M = -3.0e38f, L = 0.f, A = 0.f;
  for (int c = 0; c < NCH; ++c) {
    const float* pp = part + (h * NCH + c) * PARTSZ;
    const float mc = pp[q];
    const float lc = pp[32 + q];
    const float ac = pp[64 + q * HDIM + d];
    if (mc > M) {
      const float f = __expf(M - mc);
      A = A * f + ac;
      L = L * f + lc;
      M = mc;
    } else {
      const float e = __expf(mc - M);
      A += e * ac;
      L += e * lc;
    }
  }
  Am[q * NEMBD + h * HDIM + d] = A / L;
}

// ---------------------------------------------------------------------------
extern "C" void kernel_launch(void* const* d_in, const int* in_sizes, int n_in,
                              void* d_out, int out_size, void* d_ws, size_t ws_size,
                              hipStream_t stream) {
  const float* x      = (const float*)d_in[0];  // [32,1024]
  const float* lpast  = (const float*)d_in[1];  // [2,16,32768,64]
  const float* w_attn = (const float*)d_in[2];  // [1024,3072]
  const float* b_attn = (const float*)d_in[3];  // [3072]
  const float* w_proj = (const float*)d_in[4];  // [1024,1024]
  const float* b_proj = (const float*)d_in[5];  // [1024]
  // d_in[6]=seq_len, d_in[7]=past_len (scalars; dims are compile-time here)

  float* out     = (float*)d_out;               // [32,1024]
  float* present = out + SEQ * NEMBD;           // [2,16,32800,64]

  float* ws   = (float*)d_ws;
  float* qkv  = ws;                             // 32*3072       =   98304 f
  float* Am   = ws + 98304;                     // 32*1024       =   32768 f
  float* part = ws + 131072;                    // 16*65*2112    = 2196480 f

  // 1) qkv = x @ w_attn + b_attn
  gemm32_bias_wmma<<<3 * NEMBD / 16, 64, 0, stream>>>(x, w_attn, b_attn, qkv,
                                                      NEMBD, 3 * NEMBD);
  // 2) fused present-copy + split-K flash attention partials (async-LDS path)
  attn_chunk<<<NHEAD * NCH / 2, 64, 0, stream>>>(lpast, qkv, present, part);
  // 3) merge partials -> merged heads [32,1024]
  reduce_attn<<<NHEAD * SEQ, 64, 0, stream>>>(part, Am);
  // 4) out = Am @ w_proj + b_proj
  gemm32_bias_wmma<<<NEMBD / 16, 64, 0, stream>>>(Am, w_proj, b_proj, out,
                                                  NEMBD, NEMBD);
}
